// EdgeClassifierGNN_57870389346679
// MI455X (gfx1250) — matrix operand order repaired and checked by reading, not compiled, not run on hardware
//
#include <hip/hip_runtime.h>
#include <cstdint>

// Problem sizes (exact, all tile-divisible: 100000/16, 500000/16, 128/16 OK)
#define NN 100000   // nodes
#define ND 64       // node dim
#define NH 128      // hidden
#define EF 16       // edge feat
#define NE 1600000  // edges
#define NL 500000   // labeled edges
#define KPAD 288    // 2*NH + EF padded to multiple of 32
#define DEC_TILES 10         // m-tiles per decode block
#define DEC_BLOCKS 3125      // 3125 * 10 * 16 = 500000

typedef __bf16 bf16;
typedef __attribute__((ext_vector_type(8)))  bf16  v8bf;
typedef __attribute__((ext_vector_type(16))) bf16  v16bf;
typedef __attribute__((ext_vector_type(8)))  float v8f;

__device__ __forceinline__ bf16 f2bf(float f) {
  unsigned u = __float_as_uint(f);
  u = (u + 0x7FFFu + ((u >> 16) & 1u)) >> 16;          // round-to-nearest-even
  unsigned short s = (unsigned short)u;
  return __builtin_bit_cast(bf16, s);
}
__device__ __forceinline__ float bf2f(bf16 b) {
  unsigned short s = __builtin_bit_cast(unsigned short, b);
  return __uint_as_float(((unsigned)s) << 16);
}

// A-matrix 16x32 fragment (ISA 7.12.2): lane = row (M=lane&15), half = lane>>4.
// half 0 holds K {0..7, 16..23}, half 1 holds K {8..15, 24..31}  -> two 16B chunks.
__device__ __forceinline__ v16bf load_fragA(const bf16* rowptr, int ks, int half) {
  const v8bf lo = *reinterpret_cast<const v8bf*>(rowptr + ks * 32 + half * 8);
  const v8bf hi = *reinterpret_cast<const v8bf*>(rowptr + ks * 32 + half * 8 + 16);
  v16bf r;
#pragma unroll
  for (int i = 0; i < 8; ++i) { r[i] = lo[i]; r[i + 8] = hi[i]; }
  return r;
}
// B-matrix 32x16 fragment: lane = column (N=lane&15); half 0 holds K {0..7, 16..23},
// half 1 holds K {8..15, 24..31} (same as A) -> two 16B chunks.
// B = W^T, so column n of B is row n of W (row-major, contiguous).
__device__ __forceinline__ v16bf load_fragB(const bf16* colptr, int ks, int half) {
  const v8bf lo = *reinterpret_cast<const v8bf*>(colptr + ks * 32 + half * 8);
  const v8bf hi = *reinterpret_cast<const v8bf*>(colptr + ks * 32 + half * 8 + 16);
  v16bf r;
#pragma unroll
  for (int i = 0; i < 8; ++i) { r[i] = lo[i]; r[i + 8] = hi[i]; }
  return r;
}

// ---------------- utility kernels ----------------
__global__ void k_zero_f32(float* __restrict__ p, int n) {
  int t = blockIdx.x * 256 + threadIdx.x;
  if (t < n) p[t] = 0.f;
}
__global__ void k_f32_to_bf16(const float* __restrict__ s, bf16* __restrict__ d, int n) {
  int t = blockIdx.x * 256 + threadIdx.x;
  if (t < n) d[t] = f2bf(s[t]);
}
__global__ void k_pad_wc1(const float* __restrict__ src, bf16* __restrict__ dst) {
  int t = blockIdx.x * 256 + threadIdx.x;
  if (t >= NH * KPAD) return;
  int row = t / KPAD, c = t - row * KPAD;
  float v = (c < 2 * NH + EF) ? src[row * (2 * NH + EF) + c] : 0.f;
  dst[t] = f2bf(v);
}

// ---------------- segment-sum scatters (4 channels/thread, vectorized reads) ----------------
__global__ void k_scatter1(const float* __restrict__ x, const int* __restrict__ ei,
                           float* __restrict__ agg, float* __restrict__ cnt) {
  int t = blockIdx.x * 256 + threadIdx.x;            // NE*16 threads
  if (t >= NE * (ND / 4)) return;
  int e = t >> 4, q = (t & 15) << 2;
  int src = ei[e], dst = ei[NE + e];
  const float4 v = *reinterpret_cast<const float4*>(x + (size_t)src * ND + q);
  float* a = agg + (size_t)dst * ND + q;
  atomicAdd(a + 0, v.x); atomicAdd(a + 1, v.y);
  atomicAdd(a + 2, v.z); atomicAdd(a + 3, v.w);
  if (q == 0) atomicAdd(&cnt[dst], 1.0f);
}
__global__ void k_scatter2(const bf16* __restrict__ h1, const int* __restrict__ ei,
                           float* __restrict__ agg) {
  int t = blockIdx.x * 256 + threadIdx.x;            // NE*32 threads
  if (t >= NE * (NH / 4)) return;
  int e = t >> 5, q = (t & 31) << 2;
  int src = ei[e], dst = ei[NE + e];
  const bf16* hp = h1 + (size_t)src * NH + q;
  float* a = agg + (size_t)dst * NH + q;
  atomicAdd(a + 0, bf2f(hp[0])); atomicAdd(a + 1, bf2f(hp[1]));
  atomicAdd(a + 2, bf2f(hp[2])); atomicAdd(a + 3, bf2f(hp[3]));
}
__global__ void k_mean(const float* __restrict__ agg, const float* __restrict__ cnt,
                       bf16* __restrict__ mean, int dim_log2, int n) {
  int t = blockIdx.x * 256 + threadIdx.x;
  if (t >= n) return;
  float c = cnt[t >> dim_log2];
  mean[t] = f2bf(agg[t] / fmaxf(c, 1.0f));
}

// ---------------- SAGE layer: out = relu?(mean@W_l^T + b + x@W_r^T), bf16 out ----------------
__global__ void __launch_bounds__(256)
k_sage_gemm(const bf16* __restrict__ A1, const bf16* __restrict__ W1,
            const bf16* __restrict__ A2, const bf16* __restrict__ W2,
            const float* __restrict__ bias, bf16* __restrict__ out,
            int K, int do_relu) {
  __shared__ __align__(16) bf16 sA1[16 * 128];
  __shared__ __align__(16) bf16 sA2[16 * 128];
  const int m0 = blockIdx.x * 16;
  const int tid = threadIdx.x;
  const int nvec = 2 * K;                            // 16*K bf16 = 2K uint4
  const uint4* g1 = reinterpret_cast<const uint4*>(A1 + (size_t)m0 * K);
  const uint4* g2 = reinterpret_cast<const uint4*>(A2 + (size_t)m0 * K);
  for (int t = tid; t < nvec; t += 256) {
    reinterpret_cast<uint4*>(sA1)[t] = g1[t];
    reinterpret_cast<uint4*>(sA2)[t] = g2[t];
  }
  __syncthreads();

  const int lane = tid & 31, wave = tid >> 5;
  const int half = lane >> 4, idx = lane & 15;
  const int col = wave * 16 + idx;                   // output channel

  v8f acc = {};
  const bf16* a1row = sA1 + idx * K;
  const bf16* a2row = sA2 + idx * K;
  const bf16* w1row = W1 + (size_t)col * K;
  const bf16* w2row = W2 + (size_t)col * K;
  const int ksteps = K >> 5;
  for (int ks = 0; ks < ksteps; ++ks) {
    v16bf a = load_fragA(a1row, ks, half);
    v16bf b = load_fragB(w1row, ks, half);
    acc = __builtin_amdgcn_wmma_f32_16x16x32_bf16(false, a, false, b, (short)0, acc, false, false);
  }
  for (int ks = 0; ks < ksteps; ++ks) {
    v16bf a = load_fragA(a2row, ks, half);
    v16bf b = load_fragB(w2row, ks, half);
    acc = __builtin_amdgcn_wmma_f32_16x16x32_bf16(false, a, false, b, (short)0, acc, false, false);
  }
  const float bv = bias[col];
  const int rb = m0 + half * 8;
#pragma unroll
  for (int r = 0; r < 8; ++r) {
    float v = acc[r] + bv;
    if (do_relu) v = fmaxf(v, 0.f);
    out[(size_t)(rb + r) * NH + col] = f2bf(v);
  }
}

// ---------------- decode: z = concat(h_src,h_dst,attr)@wc1^T + bc1, + BN stats ----------------
// Persistent blocks: DEC_TILES 16-row tiles each; wc1 B-fragments held in registers;
// gathered rows moved global->LDS with async b128 copies, double-buffered so the
// gather of tile i+1 overlaps the WMMAs of tile i (ASYNCcnt + barrier sync).
__global__ void __launch_bounds__(256)
k_decode(const bf16* __restrict__ h2, const float* __restrict__ edge_attr,
         const int* __restrict__ eli, const bf16* __restrict__ wc1b,
         const float* __restrict__ bc1, float* __restrict__ z,
         float* __restrict__ stats) {
  __shared__ __align__(16) bf16 sA[2][16 * KPAD];
  const int tid = threadIdx.x;
  const int lane = tid & 31, wave = tid >> 5;
  const int half = lane >> 4, idx = lane & 15;
  const int col = wave * 16 + idx;

  // persistent B fragments for this wave's n-tile (9 k-steps, 72 VGPRs)
  v16bf bfrag[KPAD / 32];
  {
    const bf16* wrow = wc1b + (size_t)col * KPAD;
#pragma unroll
    for (int ks = 0; ks < KPAD / 32; ++ks) bfrag[ks] = load_fragB(wrow, ks, half);
  }
  const float bv = bc1[col];
  const int tile0 = blockIdx.x * DEC_TILES;

  // Build tile `it` into LDS buffer `buf`: async 16B gathers of the two h2 rows
  // per edge + VALU-converted edge_attr and zero pad.
  auto build = [&](int it, int buf) {
    const int m0 = (tile0 + it) * 16;
    for (int t = tid; t < 512; t += 256) {           // 512 16B chunks
      const int row = t >> 5, seg = (t >> 4) & 1, j = t & 15;
      const int node = eli[seg * NL + m0 + row];
      const bf16* g = h2 + (size_t)node * NH + j * 8;
      unsigned lds = (unsigned)(uintptr_t)(&sA[buf][row * KPAD + seg * NH + j * 8]);
      unsigned long long ga = (unsigned long long)(uintptr_t)g;
      asm volatile("global_load_async_to_lds_b128 %0, %1, off"
                   :: "v"(lds), "v"(ga) : "memory");
    }
    const int row = tid >> 4, c = tid & 15;          // 16x16 attr + 16x16 pad
    sA[buf][row * KPAD + 2 * NH + c] = f2bf(edge_attr[(size_t)(m0 + row) * EF + c]);
    sA[buf][row * KPAD + 2 * NH + EF + c] = f2bf(0.f);
  };

  build(0, 0);
  asm volatile("s_wait_asynccnt 0" ::: "memory");
  __syncthreads();

  float s = 0.f, q = 0.f;
  for (int it = 0; it < DEC_TILES; ++it) {
    const int buf = it & 1;
    if (it + 1 < DEC_TILES) build(it + 1, buf ^ 1);  // prefetch next tile

    const int m0 = (tile0 + it) * 16;
    v8f acc = {};
    const bf16* arow = sA[buf] + idx * KPAD;
#pragma unroll
    for (int ks = 0; ks < KPAD / 32; ++ks) {         // 9 WMMAs
      v16bf a = load_fragA(arow, ks, half);
      acc = __builtin_amdgcn_wmma_f32_16x16x32_bf16(false, a, false, bfrag[ks],
                                                    (short)0, acc, false, false);
    }
    const int rb = m0 + half * 8;
#pragma unroll
    for (int r = 0; r < 8; ++r) {
      float v = acc[r] + bv;
      z[(size_t)(rb + r) * NH + col] = v;
      s += v; q += v * v;
    }
    asm volatile("s_wait_asynccnt 0" ::: "memory");
    __syncthreads();
  }
  atomicAdd(&stats[col], s);
  atomicAdd(&stats[NH + col], q);
}

// ---------------- BN finalize: scale/shift per channel ----------------
__global__ void k_bn_finalize(float* __restrict__ stats, const float* __restrict__ g,
                              const float* __restrict__ b) {
  int c = threadIdx.x;                               // 128 threads
  const float n = (float)NL;
  float mu = stats[c] / n;
  float var = stats[NH + c] / n - mu * mu;
  float inv = rsqrtf(var + 1e-5f);
  float sc = g[c] * inv;
  stats[256 + c] = sc;                               // scale
  stats[384 + c] = b[c] - mu * sc;                   // shift
}

// ---------------- final: out = relu(z*sc+sh) . wc2 + bc2 (one wave per row) ----------------
__global__ void __launch_bounds__(256)
k_final(const float* __restrict__ z, const float* __restrict__ stats,
        const float* __restrict__ wc2, const float* __restrict__ bc2,
        float* __restrict__ out) {
  const int row = (blockIdx.x * 256 + threadIdx.x) >> 5;
  const int lane = threadIdx.x & 31;
  if (row >= NL) return;
  const float4 v = reinterpret_cast<const float4*>(z + (size_t)row * NH)[lane];
  const int c = lane * 4;
  float p = 0.f;
  p += fmaxf(v.x * stats[256 + c + 0] + stats[384 + c + 0], 0.f) * wc2[c + 0];
  p += fmaxf(v.y * stats[256 + c + 1] + stats[384 + c + 1], 0.f) * wc2[c + 1];
  p += fmaxf(v.z * stats[256 + c + 2] + stats[384 + c + 2], 0.f) * wc2[c + 2];
  p += fmaxf(v.w * stats[256 + c + 3] + stats[384 + c + 3], 0.f) * wc2[c + 3];
#pragma unroll
  for (int off = 16; off > 0; off >>= 1) p += __shfl_xor(p, off, 32);
  if (lane == 0) out[row] = p + bc2[0];
}

extern "C" void kernel_launch(void* const* d_in, const int* in_sizes, int n_in,
                              void* d_out, int out_size, void* d_ws, size_t ws_size,
                              hipStream_t stream) {
  const float* node_emb  = (const float*)d_in[0];
  const float* w1l       = (const float*)d_in[1];
  const float* b1l       = (const float*)d_in[2];
  const float* w1r       = (const float*)d_in[3];
  const float* w2l       = (const float*)d_in[4];
  const float* b2l       = (const float*)d_in[5];
  const float* w2r       = (const float*)d_in[6];
  const float* wc1       = (const float*)d_in[7];
  const float* bc1       = (const float*)d_in[8];
  const float* bn_g      = (const float*)d_in[9];
  const float* bn_b      = (const float*)d_in[10];
  const float* wc2       = (const float*)d_in[11];
  const float* bc2       = (const float*)d_in[12];
  const float* edge_attr = (const float*)d_in[13];
  const int*   ei        = (const int*)d_in[14];
  const int*   eli       = (const int*)d_in[15];
  float* out = (float*)d_out;

  char* p = (char*)d_ws;
  auto alloc = [&](size_t bytes) { char* r = p; p += (bytes + 255) & ~(size_t)255; return r; };
  bf16*  x_b    = (bf16*)alloc((size_t)NN * ND * 2);
  bf16*  mean1  = (bf16*)alloc((size_t)NN * ND * 2);
  bf16*  h1     = (bf16*)alloc((size_t)NN * NH * 2);
  bf16*  mean2  = (bf16*)alloc((size_t)NN * NH * 2);
  bf16*  h2     = (bf16*)alloc((size_t)NN * NH * 2);
  bf16*  w1l_b  = (bf16*)alloc((size_t)NH * ND * 2);
  bf16*  w1r_b  = (bf16*)alloc((size_t)NH * ND * 2);
  bf16*  w2l_b  = (bf16*)alloc((size_t)NH * NH * 2);
  bf16*  w2r_b  = (bf16*)alloc((size_t)NH * NH * 2);
  bf16*  wc1_b  = (bf16*)alloc((size_t)NH * KPAD * 2);
  float* stats  = (float*)alloc(512 * 4);
  float* cnt    = (float*)alloc((size_t)NN * 4);
  float* agg1   = (float*)alloc((size_t)NN * ND * 4);
  float* agg2   = (float*)alloc((size_t)NN * NH * 4);
  float* z      = (float*)alloc((size_t)NL * NH * 4);

  const dim3 B(256);
  auto gr = [](int n) { return dim3((unsigned)((n + 255) / 256)); };

  // zero accumulators (kernel-based; graph-capture safe)
  k_zero_f32<<<gr(NN * ND), B, 0, stream>>>(agg1, NN * ND);
  k_zero_f32<<<gr(NN * NH), B, 0, stream>>>(agg2, NN * NH);
  k_zero_f32<<<gr(NN), B, 0, stream>>>(cnt, NN);
  k_zero_f32<<<gr(512), B, 0, stream>>>(stats, 512);

  // bf16 conversions
  k_f32_to_bf16<<<gr(NN * ND), B, 0, stream>>>(node_emb, x_b, NN * ND);
  k_f32_to_bf16<<<gr(NH * ND), B, 0, stream>>>(w1l, w1l_b, NH * ND);
  k_f32_to_bf16<<<gr(NH * ND), B, 0, stream>>>(w1r, w1r_b, NH * ND);
  k_f32_to_bf16<<<gr(NH * NH), B, 0, stream>>>(w2l, w2l_b, NH * NH);
  k_f32_to_bf16<<<gr(NH * NH), B, 0, stream>>>(w2r, w2r_b, NH * NH);
  k_pad_wc1<<<gr(NH * KPAD), B, 0, stream>>>(wc1, wc1_b);

  // layer 1
  k_scatter1<<<gr(NE * (ND / 4)), B, 0, stream>>>(node_emb, ei, agg1, cnt);
  k_mean<<<gr(NN * ND), B, 0, stream>>>(agg1, cnt, mean1, 6, NN * ND);
  k_sage_gemm<<<dim3(NN / 16), B, 0, stream>>>(mean1, w1l_b, x_b, w1r_b, b1l, h1, ND, 1);

  // layer 2
  k_scatter2<<<gr(NE * (NH / 4)), B, 0, stream>>>(h1, ei, agg2);
  k_mean<<<gr(NN * NH), B, 0, stream>>>(agg2, cnt, mean2, 7, NN * NH);
  k_sage_gemm<<<dim3(NN / 16), B, 0, stream>>>(mean2, w2l_b, h1, w2r_b, b2l, h2, NH, 0);

  // decode + BN + head
  k_decode<<<dim3(DEC_BLOCKS), B, 0, stream>>>(h2, edge_attr, eli, wc1_b, bc1, z, stats);
  k_bn_finalize<<<dim3(1), dim3(128), 0, stream>>>(stats, bn_g, bn_b);
  k_final<<<dim3(NL * 32 / 256), B, 0, stream>>>(z, stats, wc2, bc2, out);
}
